// MERG_3633542332663
// MI455X (gfx1250) — matrix operand order repaired and checked by reading, not compiled, not run on hardware
//
#include <hip/hip_runtime.h>
#include <hip/hip_bf16.h>

// ---------------------------------------------------------------------------
// Problem constants (from reference)
// ---------------------------------------------------------------------------
#define NN      8192
#define EE      65536
#define IND     128
#define HID     64
#define BN_EPS  1e-5f

typedef __attribute__((ext_vector_type(16))) __bf16 v16bf;
typedef __attribute__((ext_vector_type(8)))  float  v8f;

#if __has_builtin(__builtin_amdgcn_global_load_async_to_lds_b128)
#define HAVE_ASYNC_LDS 1
// Pointer types required by the builtin (per hipcc diagnostic):
//   param 0: int4 vector in AS1 (global), param 1: int4 vector in AS3 (LDS)
typedef int v4i_vs __attribute__((vector_size(16)));
typedef __attribute__((address_space(1))) v4i_vs* as1_v4i_p;
typedef __attribute__((address_space(3))) v4i_vs* as3_v4i_p;
#else
#define HAVE_ASYNC_LDS 0
#endif

// Build a v16bf A/B fragment from two 16-byte chunks of bf16 memory/LDS.
__device__ __forceinline__ v16bf frag_from2(const __bf16* lo, const __bf16* hi) {
    union { uint4 u[2]; v16bf v; } cv;
    cv.u[0] = *(const uint4*)lo;
    cv.u[1] = *(const uint4*)hi;
    return cv.v;
}

// Convert 16 contiguous f32 to a v16bf fragment.
__device__ __forceinline__ v16bf frag_cvt16(const float* p) {
    v16bf r;
#pragma unroll
    for (int i = 0; i < 16; ++i) r[i] = (__bf16)p[i];
    return r;
}

// ---------------------------------------------------------------------------
// K0: fold all small weights.
//   scale[c] = gamma/sqrt(var+eps)
//   tb[c]    = scale*(ep2_b + ep3_b + conv_b*rowsum(ep2_w)) + (beta - mean*scale)
//   ep3s     = scale[i]*ep3_w[i,j]
//   Bb[i,d]  = sum_j ep3s[i,j]*proj1_b[j*64+d]        (bias part of A@p2)
//   W0s/W1s  = scale-folded (conv ⊛ ep2_w) matrices, [64 x 128]
// ---------------------------------------------------------------------------
__global__ void k_prep(const float* proj1_b, const float* conv_w, const float* conv_b,
                       const float* ep2_w, const float* ep2_b,
                       const float* ep3_w, const float* ep3_b,
                       const float* gamma, const float* beta,
                       const float* mean,  const float* var,
                       float* scale, float* tb, float* ep3s, float* Bb,
                       float* W0s, float* W1s) {
    int t = threadIdx.x;
    if (t < HID) {
        float sc = gamma[t] * rsqrtf(var[t] + BN_EPS);
        scale[t] = sc;
        float rs = 0.f;
        for (int w = 0; w < IND; ++w) rs += ep2_w[t * IND + w];
        tb[t] = sc * (ep2_b[t] + ep3_b[t] + conv_b[0] * rs) + (beta[t] - mean[t] * sc);
    }
    __syncthreads();
    for (int idx = t; idx < HID * HID; idx += blockDim.x)
        ep3s[idx] = scale[idx >> 6] * ep3_w[idx];
    __syncthreads();
    for (int idx = t; idx < HID * HID; idx += blockDim.x) {
        int i = idx >> 6, d = idx & 63;
        float a = 0.f;
        for (int j = 0; j < HID; ++j) a += ep3s[i * HID + j] * proj1_b[j * HID + d];
        Bb[idx] = a;
    }
    for (int idx = t; idx < HID * IND; idx += blockDim.x) {
        int j = idx >> 7, u = idx & 127;
        float a0 = 0.f, a1 = 0.f;
#pragma unroll
        for (int k = 0; k < 3; ++k) {
            int w = u + 1 - k;
            if (w >= 0 && w < IND) {
                float e = ep2_w[j * IND + w];
                a0 += conv_w[k] * e;       // in-channel 0 -> src
                a1 += conv_w[3 + k] * e;   // in-channel 1 -> dst
            }
        }
        float sc = scale[j];
        W0s[idx] = sc * a0;
        W1s[idx] = sc * a1;
    }
}

// ---------------------------------------------------------------------------
// K1: h (f32) -> bf16 copy for the WMMA GEMM.
// ---------------------------------------------------------------------------
__global__ void k_h2bf(const float* h, __bf16* hb) {
    int i = blockIdx.x * blockDim.x + threadIdx.x;
    if (i < NN * IND) hb[i] = (__bf16)h[i];
}

// ---------------------------------------------------------------------------
// K2: proj1f[(i*64+d), c] = sum_j ep3s[i,j] * proj1_w[(j*64+d), c]  (bf16 out)
// ---------------------------------------------------------------------------
__global__ void k_proj1f(const float* proj1_w, const float* ep3s, __bf16* p1f) {
    int g = blockIdx.x * blockDim.x + threadIdx.x;      // 4096*128 elems
    if (g >= HID * HID * IND) return;
    int r = g >> 7, c = g & 127;
    int i = r >> 6, d = r & 63;
    float a = 0.f;
    for (int j = 0; j < HID; ++j)
        a += ep3s[i * HID + j] * proj1_w[(size_t)(j * HID + d) * IND + c];
    p1f[(size_t)r * IND + c] = (__bf16)a;
}

// ---------------------------------------------------------------------------
// K3: Bsc = h_bf @ proj1f^T  -> [8192 x 4096] bf16, via v_wmma_f32_16x16x32_bf16.
// One wave computes a 16x64 output strip (4 N-tiles reuse one A fragment).
// All 4 A fragments (the wave's full K=128 slice) are preloaded so the B
// stream can overlap with WMMA issue.
// ---------------------------------------------------------------------------
__global__ __launch_bounds__(128)
void k_gemm_bsc(const __bf16* __restrict__ hb, const __bf16* __restrict__ p1f,
                __bf16* __restrict__ Bsc) {
    int wave = blockIdx.x * (blockDim.x >> 5) + (threadIdx.x >> 5);
    int m_t = wave >> 6;          // 512 M-tiles of 16 rows
    int n_s = wave & 63;          // 64 N-strips of 64 cols
    int lane = threadIdx.x & 31;
    int half = lane >> 4, r = lane & 15;

    const __bf16* arow = hb + (size_t)(m_t * 16 + r) * IND;
    v16bf a[4];
#pragma unroll
    for (int ks = 0; ks < 4; ++ks) {                       // A contiguous-16 packing
        int kb = ks * 32 + half * 16;
        a[ks] = frag_from2(arow + kb, arow + kb + 8);
    }

    v8f acc0 = {}, acc1 = {}, acc2 = {}, acc3 = {};
#pragma unroll
    for (int ks = 0; ks < 4; ++ks) {                       // K = 128 = 4 x 32
        int k0 = ks * 32 + half * 16;                      // B contiguous-16 packing
        const __bf16* bbase = p1f + (size_t)(n_s * 64 + r) * IND + k0;
        v16bf b0 = frag_from2(bbase,                 bbase + 8);
        v16bf b1 = frag_from2(bbase + 16 * IND,      bbase + 16 * IND + 8);
        v16bf b2 = frag_from2(bbase + 32 * IND,      bbase + 32 * IND + 8);
        v16bf b3 = frag_from2(bbase + 48 * IND,      bbase + 48 * IND + 8);
        acc0 = __builtin_amdgcn_wmma_f32_16x16x32_bf16(false, a[ks], false, b0, (short)0, acc0, false, false);
        acc1 = __builtin_amdgcn_wmma_f32_16x16x32_bf16(false, a[ks], false, b1, (short)0, acc1, false, false);
        acc2 = __builtin_amdgcn_wmma_f32_16x16x32_bf16(false, a[ks], false, b2, (short)0, acc2, false, false);
        acc3 = __builtin_amdgcn_wmma_f32_16x16x32_bf16(false, a[ks], false, b3, (short)0, acc3, false, false);
    }
    // D layout: VGPR v -> M = v + 8*half, N = r
#pragma unroll
    for (int v = 0; v < 8; ++v) {
        size_t m = (size_t)(m_t * 16 + v + half * 8);
        size_t nb = (size_t)(n_s * 64 + r);
        Bsc[m * 4096 + nb +  0] = (__bf16)acc0[v];
        Bsc[m * 4096 + nb + 16] = (__bf16)acc1[v];
        Bsc[m * 4096 + nb + 32] = (__bf16)acc2[v];
        Bsc[m * 4096 + nb + 48] = (__bf16)acc3[v];
    }
}

// ---------------------------------------------------------------------------
// K4: per-node vectors: p2[n], L0s[n] = W0s·h, L1t[n] = W1s·h + Bb·p2[n]
// One 64-thread block per node.
// ---------------------------------------------------------------------------
__global__ __launch_bounds__(64)
void k_nodes(const float* __restrict__ h, const float* __restrict__ proj2_w,
             const float* __restrict__ proj2_b, const float* __restrict__ W0s,
             const float* __restrict__ W1s, const float* __restrict__ Bb,
             float* __restrict__ p2, float* __restrict__ L0s, float* __restrict__ L1t) {
    __shared__ float hs[IND];
    __shared__ float p2s[HID];
    int n = blockIdx.x, c = threadIdx.x;
    hs[c]      = h[(size_t)n * IND + c];
    hs[c + 64] = h[(size_t)n * IND + 64 + c];
    __syncthreads();
    float a0 = 0.f, a1 = 0.f, a2 = proj2_b[c];
    const float* w0 = W0s + c * IND;
    const float* w1 = W1s + c * IND;
    const float* pw = proj2_w + c * IND;
    for (int u = 0; u < IND; ++u) {
        float hv = hs[u];
        a0 += w0[u] * hv; a1 += w1[u] * hv; a2 += pw[u] * hv;
    }
    p2s[c] = a2;
    p2[(size_t)n * HID + c] = a2;
    __syncthreads();
    float bb = 0.f;
    const float* brow = Bb + c * HID;
    for (int d = 0; d < HID; ++d) bb += brow[d] * p2s[d];
    L0s[(size_t)n * HID + c] = a0;
    L1t[(size_t)n * HID + c] = a1 + bb;
}

// ---------------------------------------------------------------------------
// Counting sort of edges by src
// ---------------------------------------------------------------------------
__global__ void k_zero(unsigned* cnt) {
    int i = blockIdx.x * blockDim.x + threadIdx.x;
    if (i < NN) cnt[i] = 0u;
}
__global__ void k_hist(const int* src, unsigned* cnt) {
    int e = blockIdx.x * blockDim.x + threadIdx.x;
    if (e < EE) atomicAdd(&cnt[src[e]], 1u);
}
__global__ __launch_bounds__(1024)
void k_scan(const unsigned* cnt, unsigned* offs, unsigned* cursor) {
    __shared__ unsigned tmp[1024];
    int t = threadIdx.x;
    unsigned loc[8], s = 0;
#pragma unroll
    for (int k = 0; k < 8; ++k) { unsigned v = cnt[t * 8 + k]; loc[k] = s; s += v; }
    tmp[t] = s;
    __syncthreads();
    for (int off = 1; off < 1024; off <<= 1) {
        unsigned v = (t >= off) ? tmp[t - off] : 0u;
        __syncthreads();
        tmp[t] += v;
        __syncthreads();
    }
    unsigned excl = (t == 0) ? 0u : tmp[t - 1];
#pragma unroll
    for (int k = 0; k < 8; ++k) {
        unsigned o = excl + loc[k];
        offs[t * 8 + k] = o;
        cursor[t * 8 + k] = o;
    }
}
__global__ void k_scatter(const int* src, unsigned* cursor, unsigned* sorted) {
    int e = blockIdx.x * blockDim.x + threadIdx.x;
    if (e < EE) {
        unsigned pos = atomicAdd(&cursor[src[e]], 1u);
        sorted[pos] = (unsigned)e;
    }
}

// ---------------------------------------------------------------------------
// K5: grouped SDDMM on WMMA. One wave per src node: stage B[s] (64x64 bf16,
// 8KB) in LDS once (via GLOBAL_LOAD_ASYNC_TO_LDS_B128 when available), then
// for each chunk of <=16 edges build P = p2[dst] as the WMMA B matrix and run
// 8x v_wmma (4 M-tiles x 2 K-steps).
//   out[e,c] = relu(L0s[s,c] + L1t[d,c] + (B[s]·p2[d])[c] + tb[c])
// ---------------------------------------------------------------------------
__global__ __launch_bounds__(32)
void k_edges(const __bf16* __restrict__ Bsc, const float* __restrict__ p2,
             const float* __restrict__ L0s, const float* __restrict__ L1t,
             const float* __restrict__ tb, const unsigned* __restrict__ cnt,
             const unsigned* __restrict__ offs, const unsigned* __restrict__ sorted,
             const int* __restrict__ dst, float* __restrict__ out) {
    int s = blockIdx.x;
    unsigned deg = cnt[s];
    if (deg == 0u) return;
    unsigned base = offs[s];

    __shared__ __align__(16) __bf16 Bs[HID * HID];   // 8 KB
    int lane = threadIdx.x;
    const __bf16* grow = Bsc + (size_t)s * 4096;

#if HAVE_ASYNC_LDS
    // CDNA5 async DMA: memory -> LDS without a VGPR bounce (ASYNCcnt-tracked).
#pragma unroll
    for (int it = 0; it < 16; ++it) {
        int idx = (it * 32 + lane) * 8;
        __builtin_amdgcn_global_load_async_to_lds_b128(
            (as1_v4i_p)(grow + idx),
            (as3_v4i_p)&Bs[idx],
            0, 0);
    }
#if __has_builtin(__builtin_amdgcn_s_wait_asynccnt)
    __builtin_amdgcn_s_wait_asynccnt(0);
#else
    asm volatile("s_wait_asynccnt 0x0" ::: "memory");
#endif
#else
#pragma unroll
    for (int it = 0; it < 16; ++it) {
        int idx = (it * 32 + lane) * 8;
        *(uint4*)&Bs[idx] = *(const uint4*)&grow[idx];
    }
#endif
    __syncthreads();

    int half = lane >> 4, r = lane & 15;
    const float* l0 = L0s + (size_t)s * HID;

    for (unsigned cs = 0; cs < deg; cs += 16) {
        unsigned slot = cs + (unsigned)r;
        bool valid = slot < deg;
        unsigned eid = sorted[base + (valid ? slot : deg - 1u)];
        int d = dst[eid];
        const float* prow = p2 + (size_t)d * HID;

        v8f acc[4] = {v8f{}, v8f{}, v8f{}, v8f{}};
#pragma unroll
        for (int ks = 0; ks < 2; ++ks) {                  // K = 64 = 2 x 32
            int k0 = ks * 32 + half * 16;                 // B: contiguous 16 K per lane
            v16bf bf = frag_cvt16(prow + k0);
            int kb = ks * 32 + half * 16;                 // A: contiguous 16 K per lane
#pragma unroll
            for (int mt = 0; mt < 4; ++mt) {
                const __bf16* ar = &Bs[(mt * 16 + r) * HID + kb];
                v16bf af = frag_from2(ar, ar + 8);
                acc[mt] = __builtin_amdgcn_wmma_f32_16x16x32_bf16(
                    false, af, false, bf, (short)0, acc[mt], false, false);
            }
        }
        // Epilogue: D element (v,lane) -> channel c = mt*16 + 8*half + v, edge slot r
        const float* l1 = L1t + (size_t)d * HID;
#pragma unroll
        for (int mt = 0; mt < 4; ++mt) {
            int c0 = mt * 16 + half * 8;
            float4 A0 = *(const float4*)(l0 + c0), A1 = *(const float4*)(l0 + c0 + 4);
            float4 B0 = *(const float4*)(l1 + c0), B1 = *(const float4*)(l1 + c0 + 4);
            float4 T0 = *(const float4*)(tb + c0), T1 = *(const float4*)(tb + c0 + 4);
            float4 o0, o1;
            o0.x = fmaxf(0.f, acc[mt][0] + A0.x + B0.x + T0.x);
            o0.y = fmaxf(0.f, acc[mt][1] + A0.y + B0.y + T0.y);
            o0.z = fmaxf(0.f, acc[mt][2] + A0.z + B0.z + T0.z);
            o0.w = fmaxf(0.f, acc[mt][3] + A0.w + B0.w + T0.w);
            o1.x = fmaxf(0.f, acc[mt][4] + A1.x + B1.x + T1.x);
            o1.y = fmaxf(0.f, acc[mt][5] + A1.y + B1.y + T1.y);
            o1.z = fmaxf(0.f, acc[mt][6] + A1.z + B1.z + T1.z);
            o1.w = fmaxf(0.f, acc[mt][7] + A1.w + B1.w + T1.w);
            if (valid) {
                float* orow = out + (size_t)eid * HID + c0;
                *(float4*)orow = o0;
                *(float4*)(orow + 4) = o1;
            }
        }
    }
}

// ---------------------------------------------------------------------------
// Host launcher
// ---------------------------------------------------------------------------
extern "C" void kernel_launch(void* const* d_in, const int* in_sizes, int n_in,
                              void* d_out, int out_size, void* d_ws, size_t ws_size,
                              hipStream_t stream) {
    const float* h       = (const float*)d_in[0];
    const int*   src     = (const int*)  d_in[1];
    const int*   dst     = (const int*)  d_in[2];
    const float* proj1_w = (const float*)d_in[3];
    const float* proj1_b = (const float*)d_in[4];
    const float* proj2_w = (const float*)d_in[5];
    const float* proj2_b = (const float*)d_in[6];
    const float* conv_w  = (const float*)d_in[7];
    const float* conv_b  = (const float*)d_in[8];
    const float* ep2_w   = (const float*)d_in[9];
    const float* ep2_b   = (const float*)d_in[10];
    const float* ep3_w   = (const float*)d_in[11];
    const float* ep3_b   = (const float*)d_in[12];
    const float* gamma   = (const float*)d_in[13];
    const float* beta    = (const float*)d_in[14];
    const float* mean    = (const float*)d_in[15];
    const float* var     = (const float*)d_in[16];
    float* out = (float*)d_out;

    // Workspace carving (256B aligned). Total ~73 MB.
    char* w = (char*)d_ws;
    auto carve = [&](size_t bytes) -> void* {
        void* p = (void*)w;
        w += (bytes + 255) & ~(size_t)255;
        return p;
    };
    float*    scale  = (float*)   carve(HID * 4);
    float*    tb     = (float*)   carve(HID * 4);
    float*    ep3s   = (float*)   carve(HID * HID * 4);
    float*    Bb     = (float*)   carve(HID * HID * 4);
    float*    W0s    = (float*)   carve(HID * IND * 4);
    float*    W1s    = (float*)   carve(HID * IND * 4);
    __bf16*   hbf    = (__bf16*)  carve((size_t)NN * IND * 2);
    __bf16*   p1f    = (__bf16*)  carve((size_t)HID * HID * IND * 2);
    float*    p2     = (float*)   carve((size_t)NN * HID * 4);
    float*    L0s    = (float*)   carve((size_t)NN * HID * 4);
    float*    L1t    = (float*)   carve((size_t)NN * HID * 4);
    __bf16*   Bsc    = (__bf16*)  carve((size_t)NN * HID * HID * 2);   // 64 MB
    unsigned* cnt    = (unsigned*)carve(NN * 4);
    unsigned* offs   = (unsigned*)carve(NN * 4);
    unsigned* cursor = (unsigned*)carve(NN * 4);
    unsigned* sorted = (unsigned*)carve(EE * 4);

    // 1) fold weights
    k_prep<<<1, 256, 0, stream>>>(proj1_b, conv_w, conv_b, ep2_w, ep2_b, ep3_w, ep3_b,
                                  gamma, beta, mean, var, scale, tb, ep3s, Bb, W0s, W1s);
    // 2) h -> bf16
    k_h2bf<<<(NN * IND + 255) / 256, 256, 0, stream>>>(h, hbf);
    // 3) fused projection weights
    k_proj1f<<<(HID * HID * IND + 255) / 256, 256, 0, stream>>>(proj1_w, ep3s, p1f);
    // 4) B = h @ proj1f^T   (WMMA bf16)
    k_gemm_bsc<<<8192, 128, 0, stream>>>(hbf, p1f, Bsc);
    // 5) per-node vectors
    k_nodes<<<NN, 64, 0, stream>>>(h, proj2_w, proj2_b, W0s, W1s, Bb, p2, L0s, L1t);
    // 6) counting sort by src
    k_zero<<<(NN + 255) / 256, 256, 0, stream>>>(cnt);
    k_hist<<<(EE + 255) / 256, 256, 0, stream>>>(src, cnt);
    k_scan<<<1, 1024, 0, stream>>>(cnt, offs, cursor);
    k_scatter<<<(EE + 255) / 256, 256, 0, stream>>>(src, cursor, sorted);
    // 7) grouped WMMA edge kernel
    k_edges<<<NN, 32, 0, stream>>>(Bsc, p2, L0s, L1t, tb, cnt, offs, sorted, dst, out);
}
